// CenterHeadTrans_28853590294658
// MI455X (gfx1250) — matrix operand order repaired and checked by reading, not compiled, not run on hardware
//
#include <hip/hip_runtime.h>
#include <hip/hip_bf16.h>
#include <math.h>

typedef _Float16 half_t;
typedef __attribute__((ext_vector_type(16))) _Float16 v16h;
typedef __attribute__((ext_vector_type(8)))  _Float16 v8h;
typedef __attribute__((ext_vector_type(8)))  float    v8f;

#define CDIV(a,b) (((a)+(b)-1)/(b))

namespace cfg {
constexpr int B = 2, CIN = 512, H = 180, W = 180, N = H * W;
constexpr int HID = 128, K = 10, P = 200, NH = 8, DH = 16, FFND = 256, HC = 64;
constexpr int BN = B * N;
constexpr float BNS = 0.999995000037f;  // 1/sqrt(1+1e-5)
}

// ---- CDNA5 async memory->LDS DMA helpers (ISA §15.18.3 / §5 flow-sync) -----
__device__ __forceinline__ void async_ld_b128(unsigned lds_off, const void* gp) {
  asm volatile("global_load_async_to_lds_b128 %0, %1, off"
               :: "v"(lds_off), "v"((unsigned long long)(size_t)gp) : "memory");
}
#define WAIT_ASYNC(n) asm volatile("s_wait_asynccnt " #n ::: "memory")

// ---------------------------------------------------------------------------
// High-intensity implicit-GEMM conv/GEMM, OCT = 128 fixed.
//   8 waves (256 thr), tile = 64 pixels x 128 channels; each wave owns a
//   32x32 macro-tile -> 4 x v_wmma_f32_16x16x32_f16 per K-step.
//   LDS double-buffered; stage kt+1 streamed by GLOBAL_LOAD_ASYNC_TO_LDS_B128
//   (ASYNCcnt) while stage kt feeds the matrix pipes. Border lanes EXEC-mask
//   the DMA and zero their LDS chunk via a normal ds store instead.
// ---------------------------------------------------------------------------
template <int KS, int CINT, bool RELU_BN, bool BIAS>
__global__ __launch_bounds__(256) void conv_wmma64(
    const half_t* __restrict__ xin, const half_t* __restrict__ wp,
    const float* __restrict__ g_, const float* __restrict__ b_,
    float* __restrict__ out32, half_t* __restrict__ out16,
    int HH, int WW) {
  constexpr int KK = KS * KS * CINT;
  constexpr int KT = KK / 32;
  constexpr int A_BUF = 64 * 40;    // halves per A stage (stride 40)
  constexpr int B_BUF = 128 * 40;   // halves per B stage
  const int PC = HH * WW;
  const int tid = threadIdx.x;
  const int wave = tid >> 5, lane = tid & 31;
  const int wm = wave & 1, wn = wave >> 1;       // 2 x 4 wave grid
  const int g = lane >> 4, mloc = lane & 15;
  const int b = blockIdx.y;
  const int tile0 = blockIdx.x * 64;

  __shared__ __align__(16) half_t As[2 * A_BUF];
  __shared__ __align__(16) half_t Bs[2 * B_BUF];

  v8f acc00 = {}, acc01 = {}, acc10 = {}, acc11 = {};
  const long xbase = (long)b * PC;

  auto issue_stage = [&](int kt, int buf) {
    const int s = (kt * 32) / CINT;
    const int c0 = kt * 32 - s * CINT;
    const int dy = s / KS - KS / 2, dx = s % KS - KS / 2;
    // A: one 16B chunk per thread (64 pixels x 4 chunks)
    {
      int pixl = tid >> 2, q = tid & 3;
      int p = tile0 + pixl;
      half_t* lp = As + buf * A_BUF + pixl * 40 + q * 8;
      bool valid = false;
      long gidx = 0;
      if (p < PC) {
        int y = p / WW, x = p - y * WW;
        int yy = y + dy, xx = x + dx;
        if (KS == 1 || (yy >= 0 && yy < HH && xx >= 0 && xx < WW)) {
          valid = true;
          gidx = (xbase + (long)yy * WW + xx) * CINT + c0 + q * 8;
        }
      }
      if (valid) {
        async_ld_b128((unsigned)(size_t)lp, xin + gidx);
      } else {
        v8h z = {};
        *(v8h*)lp = z;                 // zero-pad border / tail via DS store
      }
    }
    // B: two 16B chunks per thread (128 rows x 4 chunks, k-contiguous rows)
    for (int u = tid; u < 512; u += 256) {
      int o = u >> 2, q = u & 3;
      async_ld_b128((unsigned)(size_t)(Bs + buf * B_BUF + o * 40 + q * 8),
                    wp + (long)o * KK + kt * 32 + q * 8);
    }
  };

  issue_stage(0, 0);
  for (int kt = 0; kt < KT; ++kt) {
    const int buf = kt & 1;
    if (kt + 1 < KT) {
      issue_stage(kt + 1, buf ^ 1);
      WAIT_ASYNC(2);                  // own stage-kt DMAs done (<=3/stage issued)
    } else {
      WAIT_ASYNC(0);
    }
    __syncthreads();                  // cross-wave: everyone's stage kt landed
    const half_t* Ab = As + buf * A_BUF;
    const half_t* Bb = Bs + buf * B_BUF;
    union { v16h v; v8h h[2]; } a0, a1, b0, b1;
    const int ra = wm * 32 + mloc, rb = wn * 32 + mloc;
    a0.h[0] = *(const v8h*)(Ab + ra * 40 + g * 8);
    a0.h[1] = *(const v8h*)(Ab + ra * 40 + 16 + g * 8);
    a1.h[0] = *(const v8h*)(Ab + (ra + 16) * 40 + g * 8);
    a1.h[1] = *(const v8h*)(Ab + (ra + 16) * 40 + 16 + g * 8);
    b0.h[0] = *(const v8h*)(Bb + rb * 40 + g * 8);
    b0.h[1] = *(const v8h*)(Bb + rb * 40 + 16 + g * 8);
    b1.h[0] = *(const v8h*)(Bb + (rb + 16) * 40 + g * 8);
    b1.h[1] = *(const v8h*)(Bb + (rb + 16) * 40 + 16 + g * 8);
    acc00 = __builtin_amdgcn_wmma_f32_16x16x32_f16(false, a0.v, false, b0.v, (short)0, acc00, false, false);
    acc01 = __builtin_amdgcn_wmma_f32_16x16x32_f16(false, a0.v, false, b1.v, (short)0, acc01, false, false);
    acc10 = __builtin_amdgcn_wmma_f32_16x16x32_f16(false, a1.v, false, b0.v, (short)0, acc10, false, false);
    acc11 = __builtin_amdgcn_wmma_f32_16x16x32_f16(false, a1.v, false, b1.v, (short)0, acc11, false, false);
    __syncthreads();
  }

  for (int mi = 0; mi < 2; ++mi) {
    for (int ni = 0; ni < 2; ++ni) {
      const v8f& acc = mi == 0 ? (ni == 0 ? acc00 : acc01) : (ni == 0 ? acc10 : acc11);
      const int oc = wn * 32 + ni * 16 + mloc;
      for (int j = 0; j < 8; ++j) {
        int p = tile0 + wm * 32 + mi * 16 + g * 8 + j;
        if (p >= PC) continue;
        float val = acc[j];
        if (RELU_BN) { val = val * (g_[oc] * cfg::BNS) + b_[oc]; val = fmaxf(val, 0.f); }
        if (BIAS)    { val += b_[oc]; }
        if (out32) out32[((long)b * 128 + oc) * PC + p] = val;
        if (out16) out16[((long)(b * PC + p)) * 128 + oc] = (half_t)val;
      }
    }
  }
}

// ---------------------------------------------------------------------------
// Small-OC variant (heatmap conv, OC=16/10): 2 waves, 1 WMMA per K-step.
// ---------------------------------------------------------------------------
template <int KS, int CINT, int OCT, bool RELU_BN, bool SIG, bool BIAS>
__global__ __launch_bounds__(OCT * 4) void conv_wmma(
    const half_t* __restrict__ xin, const half_t* __restrict__ wp,
    const float* __restrict__ g_, const float* __restrict__ b_,
    float* __restrict__ out32, half_t* __restrict__ out16,
    int HH, int WW, int OCL) {
  constexpr int KK = KS * KS * CINT;
  const int PC = HH * WW;
  const int tid = threadIdx.x;
  const int nthr = OCT * 4;
  const int wave = tid >> 5, lane = tid & 31;
  const int wm = wave & 1, wn = wave >> 1;
  const int g = lane >> 4, mloc = lane & 15;
  const int b = blockIdx.y;
  const int tile0 = blockIdx.x * 32;

  __shared__ __align__(16) half_t As[32 * 40];
  __shared__ __align__(16) half_t Bs[OCT * 40];

  v8f acc = {};
  const long xbase = (long)b * PC;

  for (int kt = 0; kt < KK / 32; ++kt) {
    const int s = (kt * 32) / CINT;
    const int c0 = kt * 32 - s * CINT;
    const int dy = s / KS - KS / 2, dx = s % KS - KS / 2;
    for (int u = tid; u < 512; u += nthr) {
      int pix = u >> 4, cp = u & 15;
      int p = tile0 + pix;
      unsigned int val = 0u;
      if (p < PC) {
        int y = p / WW, x = p - y * WW;
        int yy = y + dy, xx = x + dx;
        if (yy >= 0 && yy < HH && xx >= 0 && xx < WW)
          val = *(const unsigned int*)(xin +
                ((xbase + (long)yy * WW + xx) * CINT + c0 + cp * 2));
      }
      ((unsigned int*)As)[pix * 20 + cp] = val;
    }
    for (int u = tid; u < OCT * 4; u += nthr) {
      int o = u >> 2, q = u & 3;
      v8h w8 = *(const v8h*)(wp + (long)o * KK + kt * 32 + q * 8);
      *(v8h*)(Bs + o * 40 + q * 8) = w8;
    }
    __syncthreads();
    union { v16h v; v8h h[2]; } af, bf;
    af.h[0] = *(const v8h*)(As + (wm * 16 + mloc) * 40 + g * 8);
    af.h[1] = *(const v8h*)(As + (wm * 16 + mloc) * 40 + 16 + g * 8);
    bf.h[0] = *(const v8h*)(Bs + (wn * 16 + mloc) * 40 + g * 8);
    bf.h[1] = *(const v8h*)(Bs + (wn * 16 + mloc) * 40 + 16 + g * 8);
    acc = __builtin_amdgcn_wmma_f32_16x16x32_f16(false, af.v, false, bf.v,
                                                 (short)0, acc, false, false);
    __syncthreads();
  }

  const int oc = wn * 16 + mloc;
  for (int j = 0; j < 8; ++j) {
    int p = tile0 + wm * 16 + g * 8 + j;
    if (p >= PC || oc >= OCL) continue;
    float val = acc[j];
    if (RELU_BN) { val = val * (g_[oc] * cfg::BNS) + b_[oc]; val = fmaxf(val, 0.f); }
    if (BIAS)    { val += b_[oc]; }
    if (SIG)     { val = 1.0f / (1.0f + __expf(-val)); }
    if (out32) out32[((long)b * OCL + oc) * PC + p] = val;
    if (out16) out16[((long)(b * PC + p)) * OCL + oc] = (half_t)val;
  }
}

// ---------------------------------------------------------------------------
__global__ void transpose_cast(const float* __restrict__ in, half_t* __restrict__ out,
                               int C, int Nn) {
  __shared__ float t[32][33];
  int b = blockIdx.z;
  int n0 = blockIdx.x * 32, c0 = blockIdx.y * 32;
  for (int j = threadIdx.y; j < 32; j += 8) {
    int c = c0 + j, n = n0 + threadIdx.x;
    t[j][threadIdx.x] = (c < C && n < Nn) ? in[((long)b * C + c) * Nn + n] : 0.f;
  }
  __syncthreads();
  for (int j = threadIdx.y; j < 32; j += 8) {
    int n = n0 + j, c = c0 + threadIdx.x;
    if (n < Nn && c < C) out[((long)b * Nn + n) * C + c] = (half_t)t[threadIdx.x][j];
  }
}

__global__ void pack_conv_w(const float* __restrict__ w, half_t* __restrict__ wp,
                            int OCT, int OCL, int Cin, int KKl) {
  int i = blockIdx.x * blockDim.x + threadIdx.x;
  if (i >= OCT * KKl) return;
  int o = i / KKl, k = i - o * KKl;
  float v = 0.f;
  if (o < OCL) {
    int s = k / Cin, c = k - s * Cin;
    v = w[(long)(o * Cin + c) * 9 + s];
  }
  wp[i] = (half_t)v;
}

__global__ void pack_lin_w(const float* __restrict__ w, half_t* __restrict__ wp, int n) {
  int i = blockIdx.x * blockDim.x + threadIdx.x;
  if (i < n) wp[i] = (half_t)w[i];
}

__global__ void nms_kernel(const float* __restrict__ heat, float* __restrict__ supp) {
  int i = blockIdx.x * blockDim.x + threadIdx.x;
  if (i >= cfg::B * cfg::K * cfg::N) return;
  int n = i % cfg::N;
  int y = n / cfg::W, x = n % cfg::W;
  float v = heat[i], r = 0.f;
  if (y >= 1 && y < cfg::H - 1 && x >= 1 && x < cfg::W - 1) {
    float mx = -1e30f;
    for (int dy = -1; dy <= 1; ++dy)
      for (int dx = -1; dx <= 1; ++dx)
        mx = fmaxf(mx, heat[i + dy * cfg::W + dx]);
    if (v == mx) r = v;
  }
  supp[i] = r;
}

__global__ void topk_kernel(float* __restrict__ supp, int* __restrict__ tcls,
                            int* __restrict__ tpos) {
  __shared__ float sv[1024];
  __shared__ int si[1024];
  const int M = cfg::K * cfg::N;
  int b = blockIdx.x, t = threadIdx.x;
  float* base = supp + (long)b * M;
  for (int it = 0; it < cfg::P; ++it) {
    float best = -1e30f; int bi = 0;
    for (int i = t; i < M; i += 1024) {
      float v = base[i];
      if (v > best) { best = v; bi = i; }
    }
    sv[t] = best; si[t] = bi;
    __syncthreads();
    for (int s = 512; s > 0; s >>= 1) {
      if (t < s) {
        if (sv[t + s] > sv[t] || (sv[t + s] == sv[t] && si[t + s] < si[t])) {
          sv[t] = sv[t + s]; si[t] = si[t + s];
        }
      }
      __syncthreads();
    }
    if (t == 0) {
      int idx = si[0];
      tcls[b * cfg::P + it] = idx / cfg::N;
      tpos[b * cfg::P + it] = idx % cfg::N;
      base[idx] = -1e30f;
    }
    __syncthreads();
  }
}

__global__ void qpos_kernel(const int* __restrict__ tpos, float* __restrict__ qpos) {
  int i = blockIdx.x * blockDim.x + threadIdx.x;
  if (i >= cfg::B * cfg::P) return;
  int pos = tpos[i];
  qpos[i * 2]     = (float)(pos % cfg::W) + 0.5f;
  qpos[i * 2 + 1] = (float)(pos / cfg::W) + 0.5f;
}

__global__ void gather_q(const float* __restrict__ feat, const int* __restrict__ tpos,
                         const int* __restrict__ tcls, const float* __restrict__ cw,
                         const float* __restrict__ cb, float* __restrict__ q0) {
  int i = blockIdx.x * blockDim.x + threadIdx.x;
  if (i >= cfg::B * cfg::P * 128) return;
  int c = i & 127, bp = i >> 7, b = bp / cfg::P;
  int pos = tpos[bp], cls = tcls[bp];
  q0[i] = feat[((long)b * 128 + c) * cfg::N + pos] + cw[c * cfg::K + cls] + cb[c];
}

__global__ void pe_h1(const float* __restrict__ qxy, const float* __restrict__ w1,
                      const float* __restrict__ b1, const float* __restrict__ g,
                      const float* __restrict__ bb, int rows, int Wdim,
                      float* __restrict__ out32, half_t* __restrict__ out16) {
  long i = (long)blockIdx.x * blockDim.x + threadIdx.x;
  if (i >= (long)rows * 128) return;
  int row = (int)(i >> 7), o = (int)(i & 127);
  float xx, yy;
  if (qxy) { xx = qxy[row * 2]; yy = qxy[row * 2 + 1]; }
  else { xx = (float)(row % Wdim) + 0.5f; yy = (float)(row / Wdim) + 0.5f; }
  float h = w1[o * 2] * xx + w1[o * 2 + 1] * yy + b1[o];
  h = h * (g[o] * cfg::BNS) + bb[o];
  h = fmaxf(h, 0.f);
  if (out32) out32[i] = h;
  if (out16) out16[i] = (half_t)h;
}

template <bool RELU>
__global__ void linear_small(const float* __restrict__ in, const float* __restrict__ w,
                             const float* __restrict__ bias, const float* __restrict__ res,
                             float* __restrict__ out, int Cin, int Cout) {
  __shared__ float sx[256];
  int row = blockIdx.x, o = threadIdx.x;
  for (int i = threadIdx.x; i < Cin; i += blockDim.x) sx[i] = in[(long)row * Cin + i];
  __syncthreads();
  float acc = bias ? bias[o] : 0.f;
  const float* wr = w + (long)o * Cin;
  for (int c = 0; c < Cin; ++c) acc += sx[c] * wr[c];
  if (RELU) acc = fmaxf(acc, 0.f);
  if (res) acc += res[(long)row * Cout + o];
  out[(long)row * Cout + o] = acc;
}

__global__ void ln_kernel(const float* __restrict__ in, const float* __restrict__ g,
                          const float* __restrict__ b, float* __restrict__ out) {
  __shared__ float red[128];
  int row = blockIdx.x, c = threadIdx.x;
  float v = in[row * 128 + c];
  red[c] = v;
  __syncthreads();
  for (int s = 64; s > 0; s >>= 1) { if (c < s) red[c] += red[c + s]; __syncthreads(); }
  float m = red[0] * (1.f / 128.f);
  __syncthreads();
  red[c] = (v - m) * (v - m);
  __syncthreads();
  for (int s = 64; s > 0; s >>= 1) { if (c < s) red[c] += red[c + s]; __syncthreads(); }
  float var = red[0] * (1.f / 128.f);
  out[row * 128 + c] = (v - m) * rsqrtf(var + 1e-5f) * g[c] + b[c];
}

__global__ void add_rows(const float* __restrict__ a, const float* __restrict__ b,
                         float* __restrict__ o, int n) {
  int i = blockIdx.x * blockDim.x + threadIdx.x;
  if (i < n) o[i] = a[i] + b[i];
}

__global__ void addcast_kvh(const half_t* __restrict__ feath, const half_t* __restrict__ kpeh,
                            half_t* __restrict__ kvh) {
  int i = blockIdx.x * blockDim.x + threadIdx.x;
  if (i >= cfg::BN * 128) return;
  int r = i % (cfg::N * 128);
  kvh[i] = (half_t)((float)feath[i] + (float)kpeh[r]);
}

__global__ void split_kv(const float* __restrict__ sqkv, half_t* __restrict__ kh,
                         half_t* __restrict__ vh) {
  int i = blockIdx.x * blockDim.x + threadIdx.x;
  if (i >= cfg::B * cfg::P * 128) return;
  int row = i >> 7, c = i & 127;
  kh[i] = (half_t)sqkv[row * 384 + 128 + c];
  vh[i] = (half_t)sqkv[row * 384 + 256 + c];
}

__global__ void attention(const float* __restrict__ Q, int qstride,
                          const half_t* __restrict__ Kh, const half_t* __restrict__ Vh,
                          float* __restrict__ out, int Lk) {
  int wglob = blockIdx.x * 8 + (threadIdx.x >> 5);
  int lane = threadIdx.x & 31;
  int h = wglob % cfg::NH;
  int bp = wglob / cfg::NH;
  int b = bp / cfg::P;
  float qv[16];
  const float* qp = Q + (long)bp * qstride + h * 16;
  for (int d = 0; d < 16; ++d) qv[d] = qp[d] * 0.25f;
  float m = -1e30f, l = 0.f, acc[16];
  for (int d = 0; d < 16; ++d) acc[d] = 0.f;
  for (int kk = lane; kk < Lk; kk += 32) {
    const half_t* kvec = Kh + ((long)b * Lk + kk) * 128 + h * 16;
    const half_t* vvec = Vh + ((long)b * Lk + kk) * 128 + h * 16;
    float s = 0.f;
    for (int d = 0; d < 16; ++d) s += qv[d] * (float)kvec[d];
    float mn = fmaxf(m, s);
    float eo = __expf(m - mn), e = __expf(s - mn);
    l = l * eo + e;
    for (int d = 0; d < 16; ++d) acc[d] = acc[d] * eo + e * (float)vvec[d];
    m = mn;
  }
  for (int offm = 16; offm >= 1; offm >>= 1) {
    float m2 = __shfl_xor(m, offm, 32);
    float l2 = __shfl_xor(l, offm, 32);
    float mn = fmaxf(m, m2);
    float a0 = __expf(m - mn), a1 = __expf(m2 - mn);
    l = l * a0 + l2 * a1;
    for (int d = 0; d < 16; ++d) {
      float o2 = __shfl_xor(acc[d], offm, 32);
      acc[d] = acc[d] * a0 + o2 * a1;
    }
    m = mn;
  }
  if (lane == 0) {
    float inv = 1.f / l;
    for (int d = 0; d < 16; ++d) out[(long)bp * 128 + h * 16 + d] = acc[d] * inv;
  }
}

struct HeadsArgs {
  const float* w1[6]; const float* g[6]; const float* bb[6];
  const float* w2[6]; const float* b2[6];
};

__global__ void heads_kernel(const float* __restrict__ q3, const float* __restrict__ qpos,
                             HeadsArgs ha, float* __restrict__ pred) {
  __shared__ float sq[128];
  __shared__ float sh[64];
  const int ocs[6]  = {2, 1, 3, 2, 2, 10};
  const int coff[6] = {0, 2, 3, 6, 8, 10};
  int p = blockIdx.x, b = blockIdx.y, t = threadIdx.x;
  int row = b * cfg::P + p;
  sq[t] = q3[row * 128 + t];
  __syncthreads();
  for (int hh = 0; hh < 6; ++hh) {
    if (t < 64) {
      const float* wr = ha.w1[hh] + t * 128;
      float a = 0.f;
      for (int c = 0; c < 128; ++c) a += wr[c] * sq[c];
      a = a * (ha.g[hh][t] * cfg::BNS) + ha.bb[hh][t];
      sh[t] = fmaxf(a, 0.f);
    }
    __syncthreads();
    int oc = ocs[hh];
    if (t < oc) {
      const float* wr = ha.w2[hh] + t * 64;
      float a = ha.b2[hh][t];
      for (int c = 0; c < 64; ++c) a += wr[c] * sh[c];
      if (hh == 0) a += qpos[row * 2 + t];
      pred[((long)b * 20 + coff[hh] + t) * cfg::P + p] = a;
    }
    __syncthreads();
  }
}

// ---------------------------------------------------------------------------
extern "C" void kernel_launch(void* const* d_in, const int* in_sizes, int n_in,
                              void* d_out, int out_size, void* d_ws, size_t ws_size,
                              hipStream_t stream) {
  using namespace cfg;
  (void)in_sizes; (void)n_in; (void)out_size; (void)ws_size;
  const float* x        = (const float*)d_in[0];
  const float* w_shared = (const float*)d_in[1];
  const float* hm_w1    = (const float*)d_in[2];
  const float* hm_bng   = (const float*)d_in[3];
  const float* hm_bnb   = (const float*)d_in[4];
  const float* hm_w2    = (const float*)d_in[5];
  const float* ce_w     = (const float*)d_in[6];
  const float* ce_b     = (const float*)d_in[7];
  const float* spe_w1   = (const float*)d_in[8];
  const float* spe_b1   = (const float*)d_in[9];
  const float* spe_g    = (const float*)d_in[10];
  const float* spe_bb   = (const float*)d_in[11];
  const float* spe_w2   = (const float*)d_in[12];
  const float* spe_b2   = (const float*)d_in[13];
  const float* cpe_w1   = (const float*)d_in[14];
  const float* cpe_b1   = (const float*)d_in[15];
  const float* cpe_g    = (const float*)d_in[16];
  const float* cpe_bb   = (const float*)d_in[17];
  const float* cpe_w2   = (const float*)d_in[18];
  const float* cpe_b2   = (const float*)d_in[19];
  const float* sa_inw   = (const float*)d_in[20];
  const float* sa_inb   = (const float*)d_in[21];
  const float* sa_outw  = (const float*)d_in[22];
  const float* sa_outb  = (const float*)d_in[23];
  const float* ca_inw   = (const float*)d_in[24];
  const float* ca_inb   = (const float*)d_in[25];
  const float* ca_outw  = (const float*)d_in[26];
  const float* ca_outb  = (const float*)d_in[27];
  const float* lin1w    = (const float*)d_in[28];
  const float* lin1b    = (const float*)d_in[29];
  const float* lin2w    = (const float*)d_in[30];
  const float* lin2b    = (const float*)d_in[31];
  const float* n1g = (const float*)d_in[32], *n1b = (const float*)d_in[33];
  const float* n2g = (const float*)d_in[34], *n2b = (const float*)d_in[35];
  const float* n3g = (const float*)d_in[36], *n3b = (const float*)d_in[37];

  HeadsArgs ha;
  for (int i = 0; i < 6; ++i) {
    ha.w1[i] = (const float*)d_in[38 + 5 * i];
    ha.g[i]  = (const float*)d_in[39 + 5 * i];
    ha.bb[i] = (const float*)d_in[40 + 5 * i];
    ha.w2[i] = (const float*)d_in[41 + 5 * i];
    ha.b2[i] = (const float*)d_in[42 + 5 * i];
  }

  char* ws = (char*)d_ws;
  size_t off = 0;
  auto alloc = [&](size_t bytes) -> void* {
    off = (off + 255) & ~(size_t)255;
    void* p = ws + off;
    off += bytes;
    return p;
  };
  half_t* XH    = (half_t*)alloc((size_t)BN * CIN * 2);
  half_t* WP1   = (half_t*)alloc((size_t)128 * 4608 * 2);
  half_t* WP2   = (half_t*)alloc((size_t)128 * 1152 * 2);
  half_t* WP3   = (half_t*)alloc((size_t)16 * 1152 * 2);
  half_t* WCPE2 = (half_t*)alloc((size_t)128 * 128 * 2);
  half_t* WKP   = (half_t*)alloc((size_t)128 * 128 * 2);
  half_t* WVP   = (half_t*)alloc((size_t)128 * 128 * 2);
  float*  FEAT32= (float*) alloc((size_t)B * 128 * N * 4);
  half_t* FEATH = (half_t*)alloc((size_t)BN * 128 * 2);
  half_t* HMH   = (half_t*)alloc((size_t)BN * 128 * 2);
  float*  HEAT  = (float*) alloc((size_t)B * K * N * 4);
  float*  SUPP  = (float*) alloc((size_t)B * K * N * 4);
  half_t* KPEH1 = (half_t*)alloc((size_t)N * 128 * 2);
  half_t* KPEH  = (half_t*)alloc((size_t)N * 128 * 2);
  half_t* KVH   = (half_t*)alloc((size_t)BN * 128 * 2);
  half_t* KH    = (half_t*)alloc((size_t)BN * 128 * 2);
  half_t* VH    = (half_t*)alloc((size_t)BN * 128 * 2);
  int*    TOPC  = (int*)   alloc((size_t)B * P * 4);
  int*    TOPP  = (int*)   alloc((size_t)B * P * 4);
  float*  QPOS  = (float*) alloc((size_t)B * P * 2 * 4);
  const size_t RB = (size_t)B * P * 128 * 4;
  float* Q0 = (float*)alloc(RB),  *QPEH1F = (float*)alloc(RB), *QPE = (float*)alloc(RB);
  float* QSUM = (float*)alloc(RB), *T1 = (float*)alloc(RB),    *Q1 = (float*)alloc(RB);
  float* QSUM2= (float*)alloc(RB), *QC = (float*)alloc(RB),    *ATT1 = (float*)alloc(RB);
  float* ATT2 = (float*)alloc(RB), *T2 = (float*)alloc(RB),    *Q2 = (float*)alloc(RB);
  float* T3   = (float*)alloc(RB), *Q3 = (float*)alloc(RB);
  float* SQKV = (float*)alloc((size_t)B * P * 384 * 4);
  half_t* SKH = (half_t*)alloc((size_t)B * P * 128 * 2);
  half_t* SVH = (half_t*)alloc((size_t)B * P * 128 * 2);
  float* FF1  = (float*)alloc((size_t)B * P * 256 * 4);

  const int NT64 = CDIV(N, 64);     // 507 pixel tiles / batch
  const int BNT64 = CDIV(BN, 64);   // 1013 tiles for fused-batch GEMMs
  const int NT32 = CDIV(N, 32);     // 1013 tiles for small-OC conv
  const int ROWS = B * P;           // 400 decoder rows

  transpose_cast<<<dim3(NT32, CDIV(CIN, 32), B), dim3(32, 8), 0, stream>>>(x, XH, CIN, N);
  pack_conv_w<<<CDIV(128 * 4608, 256), 256, 0, stream>>>(w_shared, WP1, 128, 128, 512, 4608);
  pack_conv_w<<<CDIV(128 * 1152, 256), 256, 0, stream>>>(hm_w1, WP2, 128, 128, 128, 1152);
  pack_conv_w<<<CDIV(16 * 1152, 256), 256, 0, stream>>>(hm_w2, WP3, 16, 10, 128, 1152);
  pack_lin_w<<<CDIV(128 * 128, 256), 256, 0, stream>>>(cpe_w2, WCPE2, 128 * 128);
  pack_lin_w<<<CDIV(128 * 128, 256), 256, 0, stream>>>(ca_inw + 128 * 128, WKP, 128 * 128);
  pack_lin_w<<<CDIV(128 * 128, 256), 256, 0, stream>>>(ca_inw + 256 * 128, WVP, 128 * 128);

  // WMMA convs: async double-buffered 64x128 tiles for the OC=128 layers.
  conv_wmma64<3, 512, false, false><<<dim3(NT64, B), 256, 0, stream>>>(
      XH, WP1, nullptr, nullptr, FEAT32, FEATH, H, W);
  conv_wmma64<3, 128, true, false><<<dim3(NT64, B), 256, 0, stream>>>(
      FEATH, WP2, hm_bng, hm_bnb, nullptr, HMH, H, W);
  conv_wmma<3, 128, 16, false, true, false><<<dim3(NT32, B), 64, 0, stream>>>(
      HMH, WP3, nullptr, nullptr, HEAT, nullptr, H, W, 10);

  nms_kernel<<<CDIV(B * K * N, 256), 256, 0, stream>>>(HEAT, SUPP);
  topk_kernel<<<B, 1024, 0, stream>>>(SUPP, TOPC, TOPP);
  qpos_kernel<<<CDIV(ROWS, 128), 128, 0, stream>>>(TOPP, QPOS);
  gather_q<<<CDIV(ROWS * 128, 256), 256, 0, stream>>>(FEAT32, TOPP, TOPC, ce_w, ce_b, Q0);

  pe_h1<<<CDIV(ROWS * 128, 256), 256, 0, stream>>>(QPOS, spe_w1, spe_b1, spe_g, spe_bb,
                                                   ROWS, W, QPEH1F, nullptr);
  linear_small<false><<<ROWS, 128, 0, stream>>>(QPEH1F, spe_w2, spe_b2, nullptr, QPE, 128, 128);
  pe_h1<<<CDIV(N * 128, 256), 256, 0, stream>>>(nullptr, cpe_w1, cpe_b1, cpe_g, cpe_bb,
                                                N, W, nullptr, KPEH1);
  conv_wmma64<1, 128, false, true><<<dim3(NT64, 1), 256, 0, stream>>>(
      KPEH1, WCPE2, nullptr, cpe_b2, nullptr, KPEH, N, 1);

  addcast_kvh<<<CDIV(BN * 128, 256), 256, 0, stream>>>(FEATH, KPEH, KVH);
  conv_wmma64<1, 128, false, true><<<dim3(BNT64, 1), 256, 0, stream>>>(
      KVH, WKP, nullptr, ca_inb + 128, nullptr, KH, BN, 1);
  conv_wmma64<1, 128, false, true><<<dim3(BNT64, 1), 256, 0, stream>>>(
      KVH, WVP, nullptr, ca_inb + 256, nullptr, VH, BN, 1);

  add_rows<<<CDIV(ROWS * 128, 256), 256, 0, stream>>>(Q0, QPE, QSUM, ROWS * 128);
  linear_small<false><<<ROWS, 384, 0, stream>>>(QSUM, sa_inw, sa_inb, nullptr, SQKV, 128, 384);
  split_kv<<<CDIV(ROWS * 128, 256), 256, 0, stream>>>(SQKV, SKH, SVH);
  attention<<<ROWS * NH / 8, 256, 0, stream>>>(SQKV, 384, SKH, SVH, ATT1, P);
  linear_small<false><<<ROWS, 128, 0, stream>>>(ATT1, sa_outw, sa_outb, Q0, T1, 128, 128);
  ln_kernel<<<ROWS, 128, 0, stream>>>(T1, n1g, n1b, Q1);

  add_rows<<<CDIV(ROWS * 128, 256), 256, 0, stream>>>(Q1, QPE, QSUM2, ROWS * 128);
  linear_small<false><<<ROWS, 128, 0, stream>>>(QSUM2, ca_inw, ca_inb, nullptr, QC, 128, 128);
  attention<<<ROWS * NH / 8, 256, 0, stream>>>(QC, 128, KH, VH, ATT2, N);
  linear_small<false><<<ROWS, 128, 0, stream>>>(ATT2, ca_outw, ca_outb, Q1, T2, 128, 128);
  ln_kernel<<<ROWS, 128, 0, stream>>>(T2, n2g, n2b, Q2);

  linear_small<true><<<ROWS, 256, 0, stream>>>(Q2, lin1w, lin1b, nullptr, FF1, 128, 256);
  linear_small<false><<<ROWS, 128, 0, stream>>>(FF1, lin2w, lin2b, Q2, T3, 256, 128);
  ln_kernel<<<ROWS, 128, 0, stream>>>(T3, n3g, n3b, Q3);

  heads_kernel<<<dim3(P, B), 128, 0, stream>>>(Q3, QPOS, ha, (float*)d_out);
}
